// SelfAttentionHead_44074954392146
// MI455X (gfx1250) — matrix operand order repaired and checked by reading, not compiled, not run on hardware
//
#include <hip/hip_runtime.h>
#include <hip/hip_bf16.h>

typedef __attribute__((ext_vector_type(16))) _Float16 v16h;
typedef __attribute__((ext_vector_type(8)))  _Float16 v8h;
typedef __attribute__((ext_vector_type(8)))  float    v8f;
typedef unsigned int __attribute__((ext_vector_type(4))) v4u;
typedef int          __attribute__((ext_vector_type(4))) v4i;
typedef int          __attribute__((ext_vector_type(8))) v8i;

#define BATCH 8
#define TLEN  2048
#define CDIM  1024
#define HDIM  64

union U16 {
    v16h v;
    v8h  g[2];
    _Float16 h[16];
};

static __device__ inline v8f wmma_f16(v16h a, v16h b, v8f c) {
    return __builtin_amdgcn_wmma_f32_16x16x32_f16(
        false, a, false, b, (short)0, c, false, false);
}

// ---------------------------------------------------------------------------
// Tensor Data Mover: 2D tile load (global -> LDS).
// data_size = 8 bytes (code 3); tensor dims == tile dims (no OOB edge).
// dim0 = row length in 8B units, dim1 = #rows, stride = row pitch in 8B units.
// D# layout per CDNA5 ISA.  6-arg builtin (clang-23).
// ---------------------------------------------------------------------------
static __device__ inline void tdm_load_to_lds(uint32_t lds_addr, uint64_t gaddr,
                                              uint32_t dim0, uint32_t dim1,
                                              uint32_t stride) {
    v4u g0;
    g0.x = 1u;                                   // count=1 (valid user D#)
    g0.y = lds_addr;                             // lds_addr (bytes)
    g0.z = (uint32_t)gaddr;                      // global_addr[31:0]
    g0.w = (uint32_t)((gaddr >> 32) & 0x01FFFFFFull) | 0x80000000u; // type=2
    v8i g1;
    g1[0] = (int)(3u << 16);                                         // data_size=8B
    g1[1] = (int)((dim0 & 0xFFFFu) << 16);                           // tensor_dim0 lo
    g1[2] = (int)(((dim0 >> 16) & 0xFFFFu) | ((dim1 & 0xFFFFu) << 16)); // td0 hi | td1 lo
    g1[3] = (int)(((dim1 >> 16) & 0xFFFFu) | ((dim0 & 0xFFFFu) << 16)); // td1 hi | tile_dim0
    g1[4] = (int)(dim1 & 0xFFFFu);                                   // tile_dim1 (tile_dim2=0)
    g1[5] = (int)stride;                                             // dim0_stride[31:0]
    g1[6] = 0;
    g1[7] = 0;
    v4i z4 = {0, 0, 0, 0};
    v8i z8 = {0, 0, 0, 0, 0, 0, 0, 0};
    __builtin_amdgcn_tensor_load_to_lds(g0, g1, z4, z4, z8, 0);
}

// ---------------------------------------------------------------------------
// Kernel 0: pack Wq/Wk/Wv into f16 WMMA B-fragment layout, one contiguous
// 12KB blob per K-chunk:  wpack[kt][m][nt][lane][j], element = W[K][n] with
//   n = nt*16 + (lane&15),  K = kt*32 + (j&8)*2 + (lane>>4)*8 + (j&7)
// ---------------------------------------------------------------------------
__global__ __launch_bounds__(256) void pack_w_kernel(
    const float* __restrict__ Wq, const float* __restrict__ Wk,
    const float* __restrict__ Wv, _Float16* __restrict__ wpack) {
    int tid = blockIdx.x * 256 + threadIdx.x;     // 0 .. 12287
    if (tid >= 3 * 32 * 4 * 32) return;
    int lane = tid & 31;
    int nt   = (tid >> 5) & 3;
    int kt   = (tid >> 7) & 31;
    int m    = tid >> 12;
    const float* W = (m == 0) ? Wq : (m == 1) ? Wk : Wv;
    int n  = nt * 16 + (lane & 15);
    int hi = lane >> 4;
    _Float16* dst = wpack + ((((size_t)kt * 3 + m) * 4 + nt) * 32 + lane) * 16;
#pragma unroll
    for (int j = 0; j < 16; ++j) {
        int K = kt * 32 + (j & 8) * 2 + hi * 8 + (j & 7);
        dst[j] = (_Float16)W[(size_t)K * HDIM + n];
    }
}

// ---------------------------------------------------------------------------
// Kernel 1: q/k/v projections.  B fragments staged to LDS by TDM (double-
// buffered, wave 0 issues); A fragments pipelined in registers.
// ---------------------------------------------------------------------------
struct XFrag { float4 f0, f1, f2, f3; };

static __device__ inline void load_x(const float* p, XFrag& xf) {
    xf.f0 = *(const float4*)(p);
    xf.f1 = *(const float4*)(p + 4);
    xf.f2 = *(const float4*)(p + 16);
    xf.f3 = *(const float4*)(p + 20);
}

__global__ __launch_bounds__(256) void qkv_proj_kernel(
    const float* __restrict__ x, const _Float16* __restrict__ wpack,
    _Float16* __restrict__ q16, _Float16* __restrict__ k16,
    _Float16* __restrict__ v16) {
    __shared__ alignas(16) _Float16 bsh[2][3 * 4 * 32 * 16];  // 2 x 12KB

    const int tid    = threadIdx.x;
    const int lane   = tid & 31;
    const int wave   = tid >> 5;
    const int lane15 = lane & 15;
    const int hi     = lane >> 4;
    const int off    = hi * 8;

    const size_t arow = (size_t)blockIdx.x * 128 + wave * 16 + lane15;
    const float* xrow = x + arow * CDIM;

    const uint32_t lds0 = (uint32_t)(uintptr_t)&bsh[0][0];
    const uint32_t lds1 = (uint32_t)(uintptr_t)&bsh[1][0];
    const uint64_t wg   = (uint64_t)(uintptr_t)wpack;

    v8f acc[3][4] = {};

    auto compute = [&](const XFrag& xf, const _Float16* bbase) {
        U16 a;
        a.h[0]  = (_Float16)xf.f0.x; a.h[1]  = (_Float16)xf.f0.y;
        a.h[2]  = (_Float16)xf.f0.z; a.h[3]  = (_Float16)xf.f0.w;
        a.h[4]  = (_Float16)xf.f1.x; a.h[5]  = (_Float16)xf.f1.y;
        a.h[6]  = (_Float16)xf.f1.z; a.h[7]  = (_Float16)xf.f1.w;
        a.h[8]  = (_Float16)xf.f2.x; a.h[9]  = (_Float16)xf.f2.y;
        a.h[10] = (_Float16)xf.f2.z; a.h[11] = (_Float16)xf.f2.w;
        a.h[12] = (_Float16)xf.f3.x; a.h[13] = (_Float16)xf.f3.y;
        a.h[14] = (_Float16)xf.f3.z; a.h[15] = (_Float16)xf.f3.w;
#pragma unroll
        for (int m = 0; m < 3; ++m) {
#pragma unroll
            for (int nt = 0; nt < 4; ++nt) {
                const _Float16* bp = bbase + ((m * 4 + nt) * 32 + lane) * 16;
                U16 b;
                b.g[0] = *(const v8h*)bp;
                b.g[1] = *(const v8h*)(bp + 8);
                acc[m][nt] = wmma_f16(a.v, b.v, acc[m][nt]);
            }
        }
    };

    XFrag xf0, xf1;
    // prologue: chunk 0 -> buffer 0
    if (tid < 32) tdm_load_to_lds(lds0, wg, 1536, 1, 1536);
    load_x(xrow + off, xf0);
    __builtin_amdgcn_s_wait_tensorcnt(0);
    __syncthreads();

    for (int kt = 0; kt < 32; kt += 2) {
        // even step: compute buf0, prefetch (kt+1) -> buf1
        {
            load_x(xrow + (kt + 1) * 32 + off, xf1);
            if (tid < 32)
                tdm_load_to_lds(lds1, wg + (uint64_t)(kt + 1) * 12288u, 1536, 1, 1536);
            compute(xf0, &bsh[0][0]);
            __builtin_amdgcn_s_wait_tensorcnt(0);
            __syncthreads();
        }
        // odd step: compute buf1, prefetch (kt+2) -> buf0
        {
            if (kt + 2 < 32) {
                load_x(xrow + (kt + 2) * 32 + off, xf0);
                if (tid < 32)
                    tdm_load_to_lds(lds0, wg + (uint64_t)(kt + 2) * 12288u, 1536, 1, 1536);
            }
            compute(xf1, &bsh[1][0]);
            __builtin_amdgcn_s_wait_tensorcnt(0);
            __syncthreads();
        }
    }

    // Store C layout: row M = r + 8*hi, col n = nt*16 + lane15
    const size_t trow0 = (size_t)blockIdx.x * 128 + wave * 16 + 8 * hi;
#pragma unroll
    for (int m = 0; m < 3; ++m) {
        _Float16* dst = (m == 0) ? q16 : (m == 1) ? k16 : v16;
#pragma unroll
        for (int r = 0; r < 8; ++r) {
            size_t t = trow0 + r;
#pragma unroll
            for (int nt = 0; nt < 4; ++nt)
                dst[t * HDIM + nt * 16 + lane15] = (_Float16)acc[m][nt][r];
        }
    }
}

// ---------------------------------------------------------------------------
// Kernel 2: flash attention (scores = K Q^T, "query" role = k rows).
// One block = 128 t-rows (8 waves x 16); s-blocks of 32, double-buffered:
// q tile via TDM (wave 0), v tile via pipelined register transpose.
// ---------------------------------------------------------------------------
__global__ __launch_bounds__(256) void attn_kernel(
    const _Float16* __restrict__ q16, const _Float16* __restrict__ k16,
    const _Float16* __restrict__ v16, float* __restrict__ out) {
    __shared__ alignas(16) _Float16 qs[2][32 * 64];   // q rows [s][h]
    __shared__ alignas(16) _Float16 vts[2][64 * 32];  // v^T    [h][s]
    __shared__ alignas(16) _Float16 ps[8][16 * 32];   // per-wave P scratch

    const int tid    = threadIdx.x;
    const int lane   = tid & 31;
    const int wave   = tid >> 5;
    const int lane15 = lane & 15;
    const int hi     = lane >> 4;
    const int off    = hi * 16;

    const int b   = blockIdx.x >> 4;
    const int rb  = (blockIdx.x & 15) * 128;
    const int wrb = rb + wave * 16;
    const size_t base = (size_t)b * TLEN;

    const int row = tid >> 3;   // 0..31  (staging row)
    const int grp = tid & 7;    // 8-half group

    const uint32_t ldsq[2] = {(uint32_t)(uintptr_t)&qs[0][0],
                              (uint32_t)(uintptr_t)&qs[1][0]};

    // K rows as A fragments, registers for the whole s loop
    U16 kA[2];
    {
        const _Float16* krow = k16 + (base + wrb + lane15) * HDIM;
#pragma unroll
        for (int hc = 0; hc < 2; ++hc) {
            kA[hc].g[0] = *(const v8h*)(krow + hc * 32 + off);
            kA[hc].g[1] = *(const v8h*)(krow + hc * 32 + 8 + off);
        }
    }

    v8f  O[4] = {};
    float m_r[8], l_r[8];
#pragma unroll
    for (int r = 0; r < 8; ++r) { m_r[r] = -3.0e38f; l_r[r] = 0.0f; }

    const int send = rb + 128;

    // ---- prologue: tile 0 into buffer 0 ----
    if (tid < 32)
        tdm_load_to_lds(ldsq[0], (uint64_t)(uintptr_t)(q16 + base * HDIM),
                        16, 32, 16);   // 8B units: 16 per row, 32 rows
    {
        v8h vv = *(const v8h*)(v16 + (base + row) * HDIM + grp * 8);
#pragma unroll
        for (int j = 0; j < 8; ++j)
            vts[0][(grp * 8 + j) * 32 + row] = vv[j];
    }
    __builtin_amdgcn_s_wait_tensorcnt(0);
    __syncthreads();

    int cur = 0;
    for (int sb = 0; sb < send; sb += 32) {
        const bool hasnext = (sb + 32 < send);
        v8h vv;
        if (hasnext) {
            if (tid < 32)
                tdm_load_to_lds(ldsq[cur ^ 1],
                                (uint64_t)(uintptr_t)(q16 + (base + sb + 32) * HDIM),
                                16, 32, 16);
            vv = *(const v8h*)(v16 + (base + sb + 32 + row) * HDIM + grp * 8);
            if (sb + 64 < send) {
                __builtin_prefetch(q16 + (base + sb + 64 + row) * HDIM + grp * 8, 0, 1);
                __builtin_prefetch(v16 + (base + sb + 64 + row) * HDIM + grp * 8, 0, 1);
            }
        }

        if (sb <= wrb + 15) {   // wave-uniform: skip fully-masked s blocks
            // ---- S[16t][32s] = k . q ----
            v8f S[2] = {};
#pragma unroll
            for (int st = 0; st < 2; ++st) {
#pragma unroll
                for (int hc = 0; hc < 2; ++hc) {
                    v16h qb = *(const v16h*)(&qs[cur][(st * 16 + lane15) * 64 +
                                                      hc * 32 + hi * 16]);
                    S[st] = wmma_f16(kA[hc].v, qb, S[st]);
                }
            }

            // ---- online softmax ----
#pragma unroll
            for (int r = 0; r < 8; ++r) {
                const int t   = wrb + r + 8 * hi;
                const int s0g = sb + lane15;
                const int s1g = sb + 16 + lane15;
                float x0 = (s0g <= t) ? S[0][r] * 0.125f : -3.0e38f;
                float x1 = (s1g <= t) ? S[1][r] * 0.125f : -3.0e38f;
                float mx = fmaxf(x0, x1);
                mx = fmaxf(mx, __shfl_xor(mx, 1, 32));
                mx = fmaxf(mx, __shfl_xor(mx, 2, 32));
                mx = fmaxf(mx, __shfl_xor(mx, 4, 32));
                mx = fmaxf(mx, __shfl_xor(mx, 8, 32));
                float nm   = fmaxf(m_r[r], mx);
                float corr = __expf(m_r[r] - nm);
                m_r[r] = nm;
                float p0 = __expf(x0 - nm);
                float p1 = __expf(x1 - nm);
                float rs = p0 + p1;
                rs += __shfl_xor(rs, 1, 32);
                rs += __shfl_xor(rs, 2, 32);
                rs += __shfl_xor(rs, 4, 32);
                rs += __shfl_xor(rs, 8, 32);
                l_r[r] = l_r[r] * corr + rs;
#pragma unroll
                for (int nt = 0; nt < 4; ++nt) O[nt][r] *= corr;
                ps[wave][(r + 8 * hi) * 32 + lane15]      = (_Float16)p0;
                ps[wave][(r + 8 * hi) * 32 + 16 + lane15] = (_Float16)p1;
            }

            // ---- P (C layout) -> A fragment via per-wave LDS round trip ----
            U16 pA;
            pA.g[0] = *(const v8h*)(&ps[wave][lane15 * 32 + off]);
            pA.g[1] = *(const v8h*)(&ps[wave][lane15 * 32 + 8 + off]);

            // ---- O += P @ V ----
#pragma unroll
            for (int nt = 0; nt < 4; ++nt) {
                v16h vb = *(const v16h*)(&vts[cur][(nt * 16 + lane15) * 32 + hi * 16]);
                O[nt] = wmma_f16(pA.v, vb, O[nt]);
            }
        }

        if (hasnext) {
#pragma unroll
            for (int j = 0; j < 8; ++j)
                vts[cur ^ 1][(grp * 8 + j) * 32 + row] = vv[j];
        }
        __builtin_amdgcn_s_wait_tensorcnt(0);
        __syncthreads();
        cur ^= 1;
    }

    // ---- epilogue ----
#pragma unroll
    for (int r = 0; r < 8; ++r) {
        float inv = 1.0f / l_r[r];
        size_t t = base + wrb + r + 8 * hi;
#pragma unroll
        for (int nt = 0; nt < 4; ++nt)
            out[t * HDIM + nt * 16 + lane15] = O[nt][r] * inv;
    }
}

// ---------------------------------------------------------------------------
extern "C" void kernel_launch(void* const* d_in, const int* in_sizes, int n_in,
                              void* d_out, int out_size, void* d_ws, size_t ws_size,
                              hipStream_t stream) {
    const float* x  = (const float*)d_in[0];
    const float* Wq = (const float*)d_in[1];
    const float* Wk = (const float*)d_in[2];
    const float* Wv = (const float*)d_in[3];
    float* out = (float*)d_out;

    _Float16* wpack = (_Float16*)d_ws;                       // 196608 halves
    _Float16* q16   = wpack + 196608;
    _Float16* k16   = q16 + (size_t)BATCH * TLEN * HDIM;
    _Float16* v16   = k16 + (size_t)BATCH * TLEN * HDIM;

    pack_w_kernel<<<48, 256, 0, stream>>>(Wq, Wk, Wv, wpack);
    qkv_proj_kernel<<<(BATCH * TLEN) / 128, 256, 0, stream>>>(x, wpack, q16, k16, v16);
    attn_kernel<<<BATCH * (TLEN / 128), 256, 0, stream>>>(q16, k16, v16, out);
    (void)in_sizes; (void)n_in; (void)out_size; (void)ws_size;
}
